// STICKYLlamaAttention_43198781063598
// MI455X (gfx1250) — matrix operand order repaired and checked by reading, not compiled, not run on hardware
//
#include <hip/hip_runtime.h>
#include <hip/hip_bf16.h>

// ---------------------------------------------------------------------------
// Trivial ext-vector types (keep unions trivially constructible)
// ---------------------------------------------------------------------------
typedef __bf16         bf16x16 __attribute__((ext_vector_type(16)));
typedef __bf16         bf16x8  __attribute__((ext_vector_type(8)));
typedef float          f32x8   __attribute__((ext_vector_type(8)));
typedef float          f32x4   __attribute__((ext_vector_type(4)));
typedef unsigned int   u32x4   __attribute__((ext_vector_type(4)));
typedef unsigned int   u32x8   __attribute__((ext_vector_type(8)));

union FragAB { bf16x16 v; u32x4 q[2]; };

__device__ __forceinline__ unsigned short f2bf_u(float x) {
  union { __bf16 b; unsigned short u; } c; c.b = (__bf16)x; return c.u;
}

#define LDT 40   // padded LDS row stride (ushorts) for 32-wide bf16 tiles

// ---------------------------------------------------------------------------
// CDNA5 async global->LDS copy (16B per lane, ASYNCcnt-tracked) + waits
// ---------------------------------------------------------------------------
__device__ __forceinline__ void async_b128(const void* g, void* l) {
  unsigned           lds = (unsigned)(size_t)l;      // flat->LDS: addr[31:0]
  unsigned long long ga  = (unsigned long long)(size_t)g;
  asm volatile("global_load_async_to_lds_b128 %0, %1, off"
               :: "v"(lds), "v"(ga) : "memory");
}

template <int N>
__device__ __forceinline__ void async_wait() {
#if __has_builtin(__builtin_amdgcn_s_wait_asynccnt)
  __builtin_amdgcn_s_wait_asynccnt(N);
#else
  asm volatile("s_wait_asynccnt %0" :: "i"(N) : "memory");
#endif
}

template <int N>
__device__ __forceinline__ void tensor_wait() {
#if __has_builtin(__builtin_amdgcn_s_wait_tensorcnt)
  __builtin_amdgcn_s_wait_tensorcnt(N);
#else
  asm volatile("s_wait_tensorcnt %0" :: "i"(N) : "memory");
#endif
}

// ---------------------------------------------------------------------------
// TDM: one-instruction DMA of a 2D bf16 tile (tile_w x tile_h, row stride in
// elements) from global to LDS, with hardware LDS padding producing the LDT=40
// row layout (pad_interval = 16 DWORDs = one 64B tile row, pad_amount = 4
// DWORDs = 16B). D# per CDNA5 ISA. TENSORcnt.
// ---------------------------------------------------------------------------
__device__ __forceinline__ void tdm_load_2d_bf16(
    unsigned lds_addr, const __bf16* g, unsigned tile_w, unsigned tile_h,
    unsigned long long row_stride, unsigned long long dim0, unsigned long long dim1)
{
  unsigned long long ga = (unsigned long long)(size_t)g;
  u32x4 g0;
  g0[0] = 1u;                                                 // count=1, user D#
  g0[1] = lds_addr;                                           // lds_addr[31:0]
  g0[2] = (unsigned)ga;                                       // global_addr lo
  g0[3] = ((unsigned)(ga >> 32) & 0x01FFFFFFu) | (2u << 30);  // addr hi | type=2
  u32x8 g1;
  g1[0] = (1u << 16)            // data_size = 1 -> 2 bytes
        | (1u << 20)            // pad_enable
        | (3u << 22)            // pad_interval = 16 DWORDs
        | (3u << 25);           // pad_amount  = 4 DWORDs
  g1[1] = ((unsigned)(dim0 & 0xFFFFu)) << 16;                       // tensor_dim0 lo
  g1[2] = ((unsigned)((dim0 >> 16) & 0xFFFFu))
        | (((unsigned)(dim1 & 0xFFFFu)) << 16);                     // dim0 hi | dim1 lo
  g1[3] = ((unsigned)((dim1 >> 16) & 0xFFFFu)) | (tile_w << 16);    // dim1 hi | tile_dim0
  g1[4] = tile_h & 0xFFFFu;                                         // tile_dim1 (dim2=0)
  g1[5] = (unsigned)(row_stride & 0xFFFFFFFFull);                   // dim0_stride lo
  g1[6] = (unsigned)((row_stride >> 32) & 0xFFFFull);               // dim0_stride hi
  g1[7] = 0u;
  asm volatile("tensor_load_to_lds %0, %1" :: "s"(g0), "s"(g1) : "memory");
}

// ---------------------------------------------------------------------------
// One-time layout/precision fixes (weights would be bf16 at rest in practice)
// ---------------------------------------------------------------------------
__global__ __launch_bounds__(256) void cvt_f32_bf16_kernel(
    const float* __restrict__ src, __bf16* __restrict__ dst, int n8)
{
  int i = blockIdx.x * blockDim.x + threadIdx.x;
  if (i >= n8) return;
  f32x4 a = *(const f32x4*)(src + (size_t)i * 8);
  f32x4 b = *(const f32x4*)(src + (size_t)i * 8 + 4);
  bf16x8 o;
  o[0] = (__bf16)a.x; o[1] = (__bf16)a.y; o[2] = (__bf16)a.z; o[3] = (__bf16)a.w;
  o[4] = (__bf16)b.x; o[5] = (__bf16)b.y; o[6] = (__bf16)b.z; o[7] = (__bf16)b.w;
  *(bf16x8*)(dst + (size_t)i * 8) = o;
}

// W[K][N] f32 -> WT[N][K] bf16 (32x32 LDS tile, coalesced both sides)
__global__ __launch_bounds__(256) void transpose_cvt_kernel(
    const float* __restrict__ W, __bf16* __restrict__ WT, int K, int N)
{
  __shared__ float t[32][33];
  const int tx = threadIdx.x & 31, ty = threadIdx.x >> 5;   // 32 x 8
  const size_t n0 = (size_t)blockIdx.x * 32;
  const size_t k0 = (size_t)blockIdx.y * 32;
  #pragma unroll
  for (int i = 0; i < 4; ++i)
    t[ty + 8 * i][tx] = W[(k0 + ty + 8 * i) * N + n0 + tx];
  __syncthreads();
  #pragma unroll
  for (int i = 0; i < 4; ++i)
    WT[(n0 + ty + 8 * i) * K + k0 + tx] = (__bf16)t[tx][ty + 8 * i];
}

// ---------------------------------------------------------------------------
// GEMM: C = A[M,K] * B[K,N] with B given transposed (BT[N][K], row-major).
// bf16 inputs, f32 WMMA accumulate. 128x128 block tile, 8 waves, wave tile
// 32x64. Tiles staged by the Tensor Data Mover (1 instruction per tile, wave 0
// issues, TENSORcnt double-buffered).
// OUT: 0 = f32 row-major, 1 = bf16 row-major, 2 = bf16 transposed [N][M].
// ---------------------------------------------------------------------------
template <int OUT>
__global__ __launch_bounds__(256) void gemm_bf16_wmma_kernel(
    const __bf16* __restrict__ A, const __bf16* __restrict__ BT,
    void* __restrict__ Cv, int M, int N, int K)
{
  __shared__ unsigned short As[2][128 * LDT];
  __shared__ unsigned short Bs[2][128 * LDT];

  const int tid  = threadIdx.x;
  const int lane = tid & 31;
  const int wid  = tid >> 5;
  const int wm   = wid & 3;                  // 4 row-waves
  const int wn   = wid >> 2;                 // 2 col-waves
  const int hi   = lane >> 4;
  const int ln   = lane & 15;
  const size_t bm = (size_t)blockIdx.y * 128;
  const size_t bn = (size_t)blockIdx.x * 128;

  f32x8 acc[2][4];
  #pragma unroll
  for (int m = 0; m < 2; ++m)
    #pragma unroll
    for (int n = 0; n < 4; ++n) acc[m][n] = {};

  auto issue = [&](int buf, int k0) {
    if (wid == 0) {
      tdm_load_2d_bf16((unsigned)(size_t)&As[buf][0], A + bm * (size_t)K + k0,
                       32u, 128u, (unsigned long long)K,
                       (unsigned long long)K, (unsigned long long)M);
      tdm_load_2d_bf16((unsigned)(size_t)&Bs[buf][0], BT + bn * (size_t)K + k0,
                       32u, 128u, (unsigned long long)K,
                       (unsigned long long)K, (unsigned long long)N);
    }
  };

  issue(0, 0);
  const int nk = K >> 5;
  for (int t = 0; t < nk; ++t) {
    const int buf = t & 1;
    if (t + 1 < nk) {
      issue(buf ^ 1, (t + 1) << 5);          // prefetch next tile pair
      if (wid == 0) tensor_wait<2>();        // in-order => current pair landed
    } else {
      if (wid == 0) tensor_wait<0>();
    }
    __syncthreads();                         // release all waves

    FragAB a[2], b[4];
    #pragma unroll
    for (int m = 0; m < 2; ++m) {
      int row = wm * 32 + m * 16 + ln;       // A: row = lane, split-K halves
      a[m].q[0] = *(const u32x4*)&As[buf][row * LDT + hi * 8];
      a[m].q[1] = *(const u32x4*)&As[buf][row * LDT + 16 + hi * 8];
    }
    #pragma unroll
    for (int n = 0; n < 4; ++n) {
      int col = wn * 64 + n * 16 + ln;       // B: col = lane, K-striped
      b[n].q[0] = *(const u32x4*)&Bs[buf][col * LDT + hi * 8];
      b[n].q[1] = *(const u32x4*)&Bs[buf][col * LDT + 16 + hi * 8];
    }
    #pragma unroll
    for (int m = 0; m < 2; ++m)
      #pragma unroll
      for (int n = 0; n < 4; ++n)
        acc[m][n] = __builtin_amdgcn_wmma_f32_16x16x32_bf16(
            false, a[m].v, false, b[n].v, (short)0, acc[m][n], false, false);
    __syncthreads();                         // readers done before next DMA
  }

  // epilogue: C/D layout row = j + 8*hi, col = ln
  #pragma unroll
  for (int m = 0; m < 2; ++m)
    #pragma unroll
    for (int n = 0; n < 4; ++n)
      #pragma unroll
      for (int j = 0; j < 8; ++j) {
        size_t row = bm + wm * 32 + m * 16 + j + hi * 8;
        size_t col = bn + wn * 64 + n * 16 + ln;
        if (OUT == 0) ((float*)Cv)[row * (size_t)N + col] = acc[m][n][j];
        else if (OUT == 1) ((__bf16*)Cv)[row * (size_t)N + col] = (__bf16)acc[m][n][j];
        else ((__bf16*)Cv)[col * (size_t)M + row] = (__bf16)acc[m][n][j];
      }
}

// ---------------------------------------------------------------------------
// llama3 frequency-smoothed RoPE, in place on bf16, one thread per pair.
// ---------------------------------------------------------------------------
__global__ void rope_llama3_kernel(__bf16* __restrict__ X, int S, int nheads)
{
  int idx = blockIdx.x * blockDim.x + threadIdx.x;
  int total = S * nheads * 64;
  if (idx >= total) return;
  int d   = idx & 63;
  int h   = (idx >> 6) % nheads;
  int pos = idx / (64 * nheads);

  float inv     = __powf(500000.0f, -((float)(2 * d) / 128.0f));
  float wavelen = 6.28318530717958647f / inv;
  float smooth  = (8192.0f / wavelen - 1.0f) * (1.0f / 3.0f);
  smooth        = fminf(fmaxf(smooth, 0.0f), 1.0f);
  float ninv    = (1.0f - smooth) * (inv * 0.125f) + smooth * inv;
  float ang     = (float)pos * ninv;
  float c = __cosf(ang), s = __sinf(ang);

  size_t base = (size_t)pos * ((size_t)nheads * 128) + (size_t)h * 128 + d;
  float x1 = (float)X[base];
  float x2 = (float)X[base + 64];
  X[base]      = (__bf16)(x1 * c - x2 * s);
  X[base + 64] = (__bf16)(x2 * c + x1 * s);
}

// ---------------------------------------------------------------------------
// Flash attention (causal, GQA 32q/8kv, HD=128). 64 query rows per block,
// 4 waves x 16-row WMMA tiles, key blocks of 32, online softmax, O through
// WMMA's C operand. Tiles staged via async-to-LDS (ASYNCcnt path); V given
// pre-transposed as VT[dim][seq].
// ---------------------------------------------------------------------------
__global__ __launch_bounds__(128) void flash_attn_wmma_kernel(
    const __bf16* __restrict__ Q,    // [2048][4096]
    const __bf16* __restrict__ Kb,   // [2048][1024]
    const __bf16* __restrict__ VT,   // [1024][2048]
    __bf16* __restrict__ Octx)       // [2048][4096]
{
  constexpr float SC = 0.08838834764831845f;   // 1/sqrt(128)
  __shared__ unsigned short Qs [64  * 136];    // [qrow][hd]
  __shared__ unsigned short Ks [32  * 136];    // [key ][hd]
  __shared__ unsigned short VTs[128 * LDT];    // [hd  ][key]
  __shared__ unsigned short Ps [4 * 16 * LDT]; // per-wave P tile

  const int tid  = threadIdx.x;
  const int lane = tid & 31;
  const int wave = tid >> 5;
  const int hi   = lane >> 4;
  const int ln   = lane & 15;
  const int head = blockIdx.y;
  const int kvh  = head >> 2;
  const int qbase = blockIdx.x * 64;

  // stage Q tile (64x128 bf16): 1024 chunks, 8 per thread, async
  #pragma unroll
  for (int i = 0; i < 8; ++i) {
    int idx = tid + i * 128;
    int row = idx >> 4;
    int ch  = (idx & 15) << 3;
    async_b128(Q + (size_t)(qbase + row) * 4096 + head * 128 + ch,
               &Qs[row * 136 + ch]);
  }

  f32x8 o[8];
  #pragma unroll
  for (int t = 0; t < 8; ++t) o[t] = {};
  float mrow[8], lrow[8];
  #pragma unroll
  for (int j = 0; j < 8; ++j) { mrow[j] = -1e30f; lrow[j] = 0.0f; }

  const int nkb = (qbase + 64) >> 5;
  for (int kb = 0; kb < nkb; ++kb) {
    const int ks = kb << 5;
    __syncthreads();                         // prior readers of Ks/VTs done
    // stage K (32x128) and VT (128x32), both row-major bf16, async
    #pragma unroll
    for (int i = 0; i < 4; ++i) {
      int idx = tid + i * 128;
      int kr  = idx >> 4;
      int ch  = (idx & 15) << 3;
      async_b128(Kb + (size_t)(ks + kr) * 1024 + kvh * 128 + ch,
                 &Ks[kr * 136 + ch]);
      int vr  = idx >> 2;
      int vc  = (idx & 3) << 3;
      async_b128(VT + (size_t)(kvh * 128 + vr) * 2048 + ks + vc,
                 &VTs[vr * LDT + vc]);
    }
    async_wait<0>();
    __syncthreads();

    // S = Q K^T : 16x32 per wave (K=128 in 4 steps of 32)
    f32x8 s[2]; s[0] = {}; s[1] = {};
    #pragma unroll
    for (int kc = 0; kc < 4; ++kc) {
      FragAB aq;
      const int arow = wave * 16 + ln;
      aq.q[0] = *(const u32x4*)&Qs[arow * 136 + kc * 32 + hi * 8];
      aq.q[1] = *(const u32x4*)&Qs[arow * 136 + kc * 32 + 16 + hi * 8];
      #pragma unroll
      for (int n = 0; n < 2; ++n) {
        FragAB bk;
        const int key = n * 16 + ln;
        bk.q[0] = *(const u32x4*)&Ks[key * 136 + kc * 32 + hi * 8];
        bk.q[1] = *(const u32x4*)&Ks[key * 136 + kc * 32 + 16 + hi * 8];
        s[n] = __builtin_amdgcn_wmma_f32_16x16x32_bf16(
            false, aq.v, false, bk.v, (short)0, s[n], false, false);
      }
    }

    // causal mask + online softmax
    float mt[8], rs[8], al[8];
    #pragma unroll
    for (int j = 0; j < 8; ++j) {
      const int qrow = qbase + wave * 16 + j + hi * 8;
      float s0 = s[0][j] * SC;
      float s1 = s[1][j] * SC;
      if (ks + ln > qrow)      s0 = -1e30f;
      if (ks + 16 + ln > qrow) s1 = -1e30f;
      s[0][j] = s0; s[1][j] = s1;
      mt[j] = fmaxf(s0, s1);
    }
    #pragma unroll
    for (int j = 0; j < 8; ++j)
      #pragma unroll
      for (int off = 1; off < 16; off <<= 1)
        mt[j] = fmaxf(mt[j], __shfl_xor(mt[j], off, 32));
    #pragma unroll
    for (int j = 0; j < 8; ++j) {
      float mn = fmaxf(mrow[j], mt[j]);
      al[j]   = __expf(mrow[j] - mn);
      mrow[j] = mn;
      float p0 = __expf(s[0][j] - mn);
      float p1 = __expf(s[1][j] - mn);
      rs[j] = p0 + p1;
      const int prow = wave * 16 + j + hi * 8;
      Ps[prow * LDT + ln]      = f2bf_u(p0);
      Ps[prow * LDT + 16 + ln] = f2bf_u(p1);
    }
    #pragma unroll
    for (int j = 0; j < 8; ++j)
      #pragma unroll
      for (int off = 1; off < 16; off <<= 1)
        rs[j] += __shfl_xor(rs[j], off, 32);
    #pragma unroll
    for (int j = 0; j < 8; ++j) lrow[j] = lrow[j] * al[j] + rs[j];
    #pragma unroll
    for (int t = 0; t < 8; ++t)
      #pragma unroll
      for (int j = 0; j < 8; ++j) o[t][j] *= al[j];

    // O += P x V
    FragAB ap;
    ap.q[0] = *(const u32x4*)&Ps[(wave * 16 + ln) * LDT + hi * 8];
    ap.q[1] = *(const u32x4*)&Ps[(wave * 16 + ln) * LDT + 16 + hi * 8];
    #pragma unroll
    for (int t = 0; t < 8; ++t) {
      FragAB bv;
      const int d = t * 16 + ln;
      bv.q[0] = *(const u32x4*)&VTs[d * LDT + hi * 8];
      bv.q[1] = *(const u32x4*)&VTs[d * LDT + 16 + hi * 8];
      o[t] = __builtin_amdgcn_wmma_f32_16x16x32_bf16(
          false, ap.v, false, bv.v, (short)0, o[t], false, false);
    }
  }

  // normalize + store bf16 context
  float inv[8];
  #pragma unroll
  for (int j = 0; j < 8; ++j) inv[j] = 1.0f / lrow[j];
  #pragma unroll
  for (int t = 0; t < 8; ++t)
    #pragma unroll
    for (int j = 0; j < 8; ++j) {
      size_t row = (size_t)qbase + wave * 16 + j + hi * 8;
      size_t col = (size_t)head * 128 + t * 16 + ln;
      Octx[row * 4096 + col] = (__bf16)(o[t][j] * inv[j]);
    }
}

// ---------------------------------------------------------------------------
// Launch. Shapes: B=1, S=2048, HID=4096, NH=32, NKV=8, HD=128.
// Workspace (bf16): Xb | WqT | WkT | WvT | WoT | Qb | Kb | VT | Ctx  (~143 MB)
// ---------------------------------------------------------------------------
extern "C" void kernel_launch(void* const* d_in, const int* in_sizes, int n_in,
                              void* d_out, int out_size, void* d_ws, size_t ws_size,
                              hipStream_t stream) {
  (void)in_sizes; (void)n_in; (void)out_size; (void)ws_size;
  const float* X  = (const float*)d_in[0];
  const float* Wq = (const float*)d_in[1];
  const float* Wk = (const float*)d_in[2];
  const float* Wv = (const float*)d_in[3];
  const float* Wo = (const float*)d_in[4];
  float* out = (float*)d_out;

  __bf16* Xb  = (__bf16*)d_ws;
  __bf16* WqT = Xb  + (size_t)2048 * 4096;
  __bf16* WkT = WqT + (size_t)4096 * 4096;
  __bf16* WvT = WkT + (size_t)1024 * 4096;
  __bf16* WoT = WvT + (size_t)1024 * 4096;
  __bf16* Qb  = WoT + (size_t)4096 * 4096;
  __bf16* Kb  = Qb  + (size_t)2048 * 4096;
  __bf16* VT  = Kb  + (size_t)2048 * 1024;
  __bf16* Ctx = VT  + (size_t)1024 * 2048;

  // precision/layout pass: X -> bf16; W -> bf16 transposed ([N][K])
  cvt_f32_bf16_kernel<<<(2048 * 4096 / 8 + 255) / 256, 256, 0, stream>>>(
      X, Xb, 2048 * 4096 / 8);
  transpose_cvt_kernel<<<dim3(4096 / 32, 4096 / 32), 256, 0, stream>>>(Wq, WqT, 4096, 4096);
  transpose_cvt_kernel<<<dim3(1024 / 32, 4096 / 32), 256, 0, stream>>>(Wk, WkT, 4096, 1024);
  transpose_cvt_kernel<<<dim3(1024 / 32, 4096 / 32), 256, 0, stream>>>(Wv, WvT, 4096, 1024);
  transpose_cvt_kernel<<<dim3(4096 / 32, 4096 / 32), 256, 0, stream>>>(Wo, WoT, 4096, 4096);

  // QKV projections (bf16 WMMA, f32 accumulate); V written pre-transposed
  gemm_bf16_wmma_kernel<1><<<dim3(4096 / 128, 2048 / 128), 256, 0, stream>>>(
      Xb, WqT, Qb, 2048, 4096, 4096);
  gemm_bf16_wmma_kernel<1><<<dim3(1024 / 128, 2048 / 128), 256, 0, stream>>>(
      Xb, WkT, Kb, 2048, 1024, 4096);
  gemm_bf16_wmma_kernel<2><<<dim3(1024 / 128, 2048 / 128), 256, 0, stream>>>(
      Xb, WvT, VT, 2048, 1024, 4096);

  // RoPE (in place, bf16)
  rope_llama3_kernel<<<(2048 * 32 * 64 + 255) / 256, 256, 0, stream>>>(Qb, 2048, 32);
  rope_llama3_kernel<<<(2048 * 8 * 64 + 255) / 256, 256, 0, stream>>>(Kb, 2048, 8);

  // causal GQA flash attention (async-to-LDS staging)
  flash_attn_wmma_kernel<<<dim3(2048 / 64, 32), 128, 0, stream>>>(Qb, Kb, VT, Ctx);

  // output projection -> f32
  gemm_bf16_wmma_kernel<0><<<dim3(4096 / 128, 2048 / 128), 256, 0, stream>>>(
      Ctx, WoT, out, 2048, 4096, 4096);
}
